// RQVAE_90683939488005
// MI455X (gfx1250) — matrix operand order repaired and checked by reading, not compiled, not run on hardware
//
#include <hip/hip_runtime.h>

// ---------------------------------------------------------------------------
// RQ-VAE forward for MI455X (gfx1250, wave32, WMMA f16 16x16x32)
// Dual-accumulator (paired n-tile) WMMA kernels for independent matrix chains.
// ---------------------------------------------------------------------------

typedef __attribute__((ext_vector_type(16))) _Float16 v16h;
typedef __attribute__((ext_vector_type(8)))  _Float16 v8h;
typedef __attribute__((ext_vector_type(8)))  float    v8f;

#define RQ_N        65536
#define RQ_IN       768
#define RQ_D        256
#define RQ_LEVELS   4
#define RQ_K        1024

static __device__ __forceinline__ v16h rq_cat16(v8h lo, v8h hi) {
  return __builtin_shufflevector(lo, hi, 0,1,2,3,4,5,6,7,8,9,10,11,12,13,14,15);
}

// ---------------------------------------------------------------------------
// utility kernels
// ---------------------------------------------------------------------------

__global__ void rq_zero_f32(float* __restrict__ p, size_t n) {
  size_t i = (size_t)blockIdx.x * blockDim.x + threadIdx.x;
  size_t stride = (size_t)gridDim.x * blockDim.x;
  for (; i < n; i += stride) p[i] = 0.0f;
}

// dst[c*R + r] = (f16) src[r*C + c]   (weights -> [OUT, K] f16 layout)
__global__ void rq_transpose_f32_to_f16(const float* __restrict__ src,
                                        _Float16* __restrict__ dst, int R, int C) {
  int i = blockIdx.x * blockDim.x + threadIdx.x;
  int total = R * C;
  if (i < total) {
    int r = i / C, c = i % C;
    dst[(size_t)c * R + r] = (_Float16)src[i];
  }
}

__global__ void rq_convert_f32_to_f16(const float* __restrict__ src,
                                      _Float16* __restrict__ dst, size_t n) {
  size_t i = (size_t)blockIdx.x * blockDim.x + threadIdx.x;
  size_t stride = (size_t)gridDim.x * blockDim.x;
  for (; i < n; i += stride) dst[i] = (_Float16)src[i];
}

// cnorm[k] = sum_d cb[k][d]^2  over all levels*codes (flat [4*1024, 256])
__global__ void rq_cnorm(const float* __restrict__ cb, float* __restrict__ cnorm) {
  int k = blockIdx.x * blockDim.x + threadIdx.x;
  if (k < RQ_LEVELS * RQ_K) {
    const float* row = cb + (size_t)k * RQ_D;
    float s = 0.0f;
    for (int d = 0; d < RQ_D; ++d) { float v = row[d]; s += v * v; }
    cnorm[k] = s;
  }
}

// ---------------------------------------------------------------------------
// LayerNorm: x [N,768] f32 -> xn [N,768] f16
// ---------------------------------------------------------------------------

__global__ void rq_layernorm(const float* __restrict__ x,
                             const float* __restrict__ g,
                             const float* __restrict__ b,
                             _Float16* __restrict__ out) {
  __shared__ float red[256];
  int row = blockIdx.x;
  int tid = threadIdx.x;
  const float* xr = x + (size_t)row * RQ_IN;
  float v0 = xr[tid], v1 = xr[tid + 256], v2 = xr[tid + 512];

  red[tid] = v0 + v1 + v2;
  __syncthreads();
  for (int off = 128; off > 0; off >>= 1) {
    if (tid < off) red[tid] += red[tid + off];
    __syncthreads();
  }
  float mu = red[0] * (1.0f / 768.0f);
  __syncthreads();

  float d0 = v0 - mu, d1 = v1 - mu, d2 = v2 - mu;
  red[tid] = d0 * d0 + d1 * d1 + d2 * d2;
  __syncthreads();
  for (int off = 128; off > 0; off >>= 1) {
    if (tid < off) red[tid] += red[tid + off];
    __syncthreads();
  }
  float rstd = rsqrtf(red[0] * (1.0f / 768.0f) + 1e-5f);

  _Float16* o = out + (size_t)row * RQ_IN;
  o[tid]       = (_Float16)(d0 * rstd * g[tid]       + b[tid]);
  o[tid + 256] = (_Float16)(d1 * rstd * g[tid + 256] + b[tid + 256]);
  o[tid + 512] = (_Float16)(d2 * rstd * g[tid + 512] + b[tid + 512]);
}

// ---------------------------------------------------------------------------
// store helper: one 16x16 f32 accumulator tile -> memory with bias / relu
// ---------------------------------------------------------------------------

template <bool RELU>
static __device__ __forceinline__
void rq_store_tile(v8f acc, const float* __restrict__ bias,
                   float* __restrict__ outF32, _Float16* __restrict__ outF16,
                   int m0, int n0, int h, int ln, int OUT) {
  float bn = bias[n0 + ln];
#pragma unroll
  for (int r = 0; r < 8; ++r) {
    int m = m0 + h * 8 + r;
    float v = acc[r] + bn;
    if (RELU) v = v > 0.0f ? v : 0.0f;
    size_t o = (size_t)m * OUT + n0 + ln;
    if (outF32) outF32[o] = v;
    if (outF16) outF16[o] = (_Float16)v;
  }
}

// ---------------------------------------------------------------------------
// Streaming WMMA GEMM (any K): D[M,OUT] = act(A[M,K] @ Bt[OUT,K]^T + bias)
// wave handles 16 rows x 32 cols (two n-tiles, two independent WMMA chains)
// block = (32,4): 4 waves cover 128 cols;  grid = (M/16, OUT/128)
// ---------------------------------------------------------------------------

template <bool RELU>
__global__ __launch_bounds__(128)
void rq_wmma_gemm_stream(const _Float16* __restrict__ A,
                         const _Float16* __restrict__ Bt,
                         const float* __restrict__ bias,
                         float* __restrict__ outF32,
                         _Float16* __restrict__ outF16,
                         int K, int OUT) {
  int lane = threadIdx.x;          // 0..31
  int h    = lane >> 4;            // lane half
  int ln   = lane & 15;
  int m0   = blockIdx.x * 16;
  int n0a  = (blockIdx.y * 4 + threadIdx.y) * 32;
  int n0b  = n0a + 16;

  const _Float16* arow  = A  + (size_t)(m0  + ln) * K;
  const _Float16* browa = Bt + (size_t)(n0a + ln) * K;
  const _Float16* browb = Bt + (size_t)(n0b + ln) * K;

  v8f acc0 = {}, acc1 = {};
  for (int k0 = 0; k0 < K; k0 += 32) {
    v8h alo  = *(const v8h*)(arow  + k0 + h * 16);
    v8h ahi  = *(const v8h*)(arow  + k0 + h * 16 + 8);
    v8h b0lo = *(const v8h*)(browa + k0 + h * 16);
    v8h b0hi = *(const v8h*)(browa + k0 + h * 16 + 8);
    v8h b1lo = *(const v8h*)(browb + k0 + h * 16);
    v8h b1hi = *(const v8h*)(browb + k0 + h * 16 + 8);
    v16h a  = rq_cat16(alo,  ahi);
    v16h b0 = rq_cat16(b0lo, b0hi);
    v16h b1 = rq_cat16(b1lo, b1hi);
    acc0 = __builtin_amdgcn_wmma_f32_16x16x32_f16(false, a, false, b0,
                                                  (short)0, acc0, false, false);
    acc1 = __builtin_amdgcn_wmma_f32_16x16x32_f16(false, a, false, b1,
                                                  (short)0, acc1, false, false);
  }

  rq_store_tile<RELU>(acc0, bias, outF32, outF16, m0, n0a, h, ln, OUT);
  rq_store_tile<RELU>(acc1, bias, outF32, outF16, m0, n0b, h, ln, OUT);
}

// ---------------------------------------------------------------------------
// A-resident WMMA GEMM (K == 256): A tile (16 x 256) held in 8 v16h regs,
// wave sweeps all OUT columns in pairs of n-tiles (dual accumulators).
// block = (32,4): 4 waves = 4 m-tiles;  grid = (M/64)
// ---------------------------------------------------------------------------

template <bool RELU>
__global__ __launch_bounds__(128)
void rq_wmma_gemm_ares(const _Float16* __restrict__ A,   // [M,256]
                       const _Float16* __restrict__ Bt,  // [OUT,256]
                       const float* __restrict__ bias,
                       float* __restrict__ outF32,
                       _Float16* __restrict__ outF16,
                       int OUT) {
  int lane = threadIdx.x;
  int h    = lane >> 4;
  int ln   = lane & 15;
  int m0   = (blockIdx.x * 4 + threadIdx.y) * 16;

  const _Float16* arow = A + (size_t)(m0 + ln) * RQ_D;
  v16h areg[8];
#pragma unroll
  for (int k = 0; k < 8; ++k) {
    v8h lo = *(const v8h*)(arow + k * 32 + h * 16);
    v8h hi = *(const v8h*)(arow + k * 32 + h * 16 + 8);
    areg[k] = rq_cat16(lo, hi);
  }

  for (int np = 0; np < OUT / 32; ++np) {
    int n0a = np * 32;
    int n0b = n0a + 16;
    const _Float16* browa = Bt + (size_t)(n0a + ln) * RQ_D;
    const _Float16* browb = Bt + (size_t)(n0b + ln) * RQ_D;
    v8f acc0 = {}, acc1 = {};
#pragma unroll
    for (int k = 0; k < 8; ++k) {
      v8h b0lo = *(const v8h*)(browa + k * 32 + h * 16);
      v8h b0hi = *(const v8h*)(browa + k * 32 + h * 16 + 8);
      v8h b1lo = *(const v8h*)(browb + k * 32 + h * 16);
      v8h b1hi = *(const v8h*)(browb + k * 32 + h * 16 + 8);
      v16h b0 = rq_cat16(b0lo, b0hi);
      v16h b1 = rq_cat16(b1lo, b1hi);
      acc0 = __builtin_amdgcn_wmma_f32_16x16x32_f16(false, areg[k], false, b0,
                                                    (short)0, acc0, false, false);
      acc1 = __builtin_amdgcn_wmma_f32_16x16x32_f16(false, areg[k], false, b1,
                                                    (short)0, acc1, false, false);
    }
    rq_store_tile<RELU>(acc0, bias, outF32, outF16, m0, n0a, h, ln, OUT);
    rq_store_tile<RELU>(acc1, bias, outF32, outF16, m0, n0b, h, ln, OUT);
  }
}

// ---------------------------------------------------------------------------
// One residual-quantization level:
//   score[m,k] = ||c_k||^2 - 2 * r_m . c_k   (WMMA, A resident, paired tiles)
//   argmin over 1024 codes, then residual/q_final/commit update (f32)
// ---------------------------------------------------------------------------

__global__ __launch_bounds__(128)
void rq_quant_level(const _Float16* __restrict__ Rf16,    // [N,256] residual f16
                    const _Float16* __restrict__ cbf16,   // [1024,256] this level
                    const float*    __restrict__ cnorm,   // [1024] this level
                    const float*    __restrict__ cbf32,   // [1024,256] this level
                    float*          __restrict__ Res,     // [N,256] residual f32
                    float*          __restrict__ Qf,      // [N,256] q_final f32
                    _Float16*       __restrict__ Qf16,    // [N,256] q_final f16
                    _Float16*       __restrict__ Rf16out, // [N,256] residual f16 (next)
                    float*          __restrict__ out_idx, // [N,4] indices (as float)
                    float*          __restrict__ commit_acc,
                    int level) {
  int lane = threadIdx.x;
  int h    = lane >> 4;
  int ln   = lane & 15;
  int m0   = (blockIdx.x * 4 + threadIdx.y) * 16;

  // A tile (16 rows x K=256) fully in registers: 8 x v16h
  const _Float16* arow = Rf16 + (size_t)(m0 + ln) * RQ_D;
  v16h areg[8];
#pragma unroll
  for (int k = 0; k < 8; ++k) {
    v8h lo = *(const v8h*)(arow + k * 32 + h * 16);
    v8h hi = *(const v8h*)(arow + k * 32 + h * 16 + 8);
    areg[k] = rq_cat16(lo, hi);
  }

  float minv[8];
  int   mini[8];
#pragma unroll
  for (int r = 0; r < 8; ++r) { minv[r] = 3.4e38f; mini[r] = 0; }

  // sweep 1024 codes as 32 pairs of 16-wide tiles (two independent WMMA chains)
  for (int nt = 0; nt < RQ_K / 32; ++nt) {
    int na = nt * 32 + ln;
    int nb = na + 16;
    const _Float16* browa = cbf16 + (size_t)na * RQ_D;
    const _Float16* browb = cbf16 + (size_t)nb * RQ_D;
    v8f acc0 = {}, acc1 = {};
#pragma unroll
    for (int k = 0; k < 8; ++k) {
      v8h b0lo = *(const v8h*)(browa + k * 32 + h * 16);
      v8h b0hi = *(const v8h*)(browa + k * 32 + h * 16 + 8);
      v8h b1lo = *(const v8h*)(browb + k * 32 + h * 16);
      v8h b1hi = *(const v8h*)(browb + k * 32 + h * 16 + 8);
      v16h b0 = rq_cat16(b0lo, b0hi);
      v16h b1 = rq_cat16(b1lo, b1hi);
      acc0 = __builtin_amdgcn_wmma_f32_16x16x32_f16(false, areg[k], false, b0,
                                                    (short)0, acc0, false, false);
      acc1 = __builtin_amdgcn_wmma_f32_16x16x32_f16(false, areg[k], false, b1,
                                                    (short)0, acc1, false, false);
    }
    float cna = cnorm[na];
    float cnb = cnorm[nb];
#pragma unroll
    for (int r = 0; r < 8; ++r) {
      float sa = cna - 2.0f * acc0[r];
      float sb = cnb - 2.0f * acc1[r];
      if (sa < minv[r] || (sa == minv[r] && na < mini[r])) { minv[r] = sa; mini[r] = na; }
      if (sb < minv[r] || (sb == minv[r] && nb < mini[r])) { minv[r] = sb; mini[r] = nb; }
    }
  }

  // reduce argmin across the 16 lanes that share each row (xor masks < 16)
#pragma unroll
  for (int r = 0; r < 8; ++r) {
#pragma unroll
    for (int off = 8; off >= 1; off >>= 1) {
      float ov = __shfl_xor(minv[r], off, 32);
      int   oi = __shfl_xor(mini[r], off, 32);
      if (ov < minv[r] || (ov == minv[r] && oi < mini[r])) { minv[r] = ov; mini[r] = oi; }
    }
  }

  // cooperative update of the 16 rows owned by this wave
  float commit_local = 0.0f;
#pragma unroll
  for (int j = 0; j < 16; ++j) {
    int idx = __shfl(mini[j & 7], (j >> 3) * 16, 32);  // broadcast row j's argmin
    int m = m0 + j;
    if (lane == 0) out_idx[(size_t)m * RQ_LEVELS + level] = (float)idx;
    const float* crow = cbf32 + (size_t)idx * RQ_D;
    size_t base = (size_t)m * RQ_D;
    for (int d = lane; d < RQ_D; d += 32) {
      float q  = crow[d];
      float nv = Res[base + d] - q;      // new residual (== r_old - q)
      Res[base + d] = nv;
      commit_local += nv * nv;
      float qf = Qf[base + d] + q;
      Qf[base + d]   = qf;
      Qf16[base + d] = (_Float16)qf;
      Rf16out[base + d] = (_Float16)nv;
    }
  }

  for (int off = 16; off >= 1; off >>= 1)
    commit_local += __shfl_xor(commit_local, off, 32);
  if (lane == 0) atomicAdd(commit_acc, commit_local);
}

// ---------------------------------------------------------------------------
// reconstruction loss: sum (recon - x)^2
// ---------------------------------------------------------------------------

__global__ void rq_recloss(const float* __restrict__ recon,
                           const float* __restrict__ x,
                           float* __restrict__ acc, size_t n) {
  __shared__ float sm[256];
  size_t i = (size_t)blockIdx.x * blockDim.x + threadIdx.x;
  size_t stride = (size_t)gridDim.x * blockDim.x;
  float s = 0.0f;
  for (; i < n; i += stride) { float d = recon[i] - x[i]; s += d * d; }
  sm[threadIdx.x] = s;
  __syncthreads();
  for (int off = 128; off > 0; off >>= 1) {
    if (threadIdx.x < off) sm[threadIdx.x] += sm[threadIdx.x + off];
    __syncthreads();
  }
  if (threadIdx.x == 0) atomicAdd(acc, sm[0]);
}

__global__ void rq_finalize(const float* __restrict__ acc, float* __restrict__ out) {
  if (threadIdx.x == 0 && blockIdx.x == 0) {
    float rec    = acc[1] / ((float)RQ_N * (float)RQ_IN);
    float commit = acc[0] / ((float)RQ_N * (float)RQ_D);
    out[0] = rec + 0.25f * commit;
  }
}

// ---------------------------------------------------------------------------
// launch
// ---------------------------------------------------------------------------

extern "C" void kernel_launch(void* const* d_in, const int* in_sizes, int n_in,
                              void* d_out, int out_size, void* d_ws, size_t ws_size,
                              hipStream_t stream) {
  (void)in_sizes; (void)n_in; (void)out_size; (void)ws_size;

  const float* x    = (const float*)d_in[0];
  const float* ln_g = (const float*)d_in[1];
  const float* ln_b = (const float*)d_in[2];
  const float* W1   = (const float*)d_in[3];   // [768,256]
  const float* b1   = (const float*)d_in[4];
  const float* W2   = (const float*)d_in[5];   // [256,256]
  const float* b2   = (const float*)d_in[6];
  const float* Wd1  = (const float*)d_in[7];   // [256,768]
  const float* bd1  = (const float*)d_in[8];
  const float* Wd2  = (const float*)d_in[9];   // [768,768]
  const float* bd2  = (const float*)d_in[10];
  const float* cb   = (const float*)d_in[11];  // [4,1024,256]
  float* out = (float*)d_out;

  char* ws = (char*)d_ws;
  // workspace layout (all sizes multiples of 256 bytes)
  const size_t o_xn   = 0;                              // [N,768] f16 (also decoder hidden)
  const size_t o_h1   = o_xn   + (size_t)RQ_N*RQ_IN*2;  // [N,256] f16 encoder hidden
  const size_t o_r16  = o_h1   + (size_t)RQ_N*RQ_D*2;   // [N,256] f16 residual
  const size_t o_res  = o_r16  + (size_t)RQ_N*RQ_D*2;   // [N,256] f32 residual
  const size_t o_qf   = o_res  + (size_t)RQ_N*RQ_D*4;   // [N,256] f32 q_final
  const size_t o_qf16 = o_qf   + (size_t)RQ_N*RQ_D*4;   // [N,256] f16 q_final
  const size_t o_w1t  = o_qf16 + (size_t)RQ_N*RQ_D*2;   // [256,768] f16
  const size_t o_w2t  = o_w1t  + (size_t)RQ_D*RQ_IN*2;  // [256,256] f16
  const size_t o_wd1t = o_w2t  + (size_t)RQ_D*RQ_D*2;   // [768,256] f16
  const size_t o_wd2t = o_wd1t + (size_t)RQ_IN*RQ_D*2;  // [768,768] f16
  const size_t o_cb16 = o_wd2t + (size_t)RQ_IN*RQ_IN*2; // [4,1024,256] f16
  const size_t o_cn   = o_cb16 + (size_t)RQ_LEVELS*RQ_K*RQ_D*2; // [4096] f32
  const size_t o_acc  = o_cn   + (size_t)RQ_LEVELS*RQ_K*4;      // [2] f32

  _Float16* xn   = (_Float16*)(ws + o_xn);
  _Float16* h1   = (_Float16*)(ws + o_h1);
  _Float16* r16  = (_Float16*)(ws + o_r16);
  float*    res  = (float*)   (ws + o_res);
  float*    qf   = (float*)   (ws + o_qf);
  _Float16* qf16 = (_Float16*)(ws + o_qf16);
  _Float16* w1t  = (_Float16*)(ws + o_w1t);
  _Float16* w2t  = (_Float16*)(ws + o_w2t);
  _Float16* wd1t = (_Float16*)(ws + o_wd1t);
  _Float16* wd2t = (_Float16*)(ws + o_wd2t);
  _Float16* cb16 = (_Float16*)(ws + o_cb16);
  float*    cn   = (float*)   (ws + o_cn);
  float*    acc  = (float*)   (ws + o_acc);

  float* recon   = out + 1;
  float* out_idx = out + 1 + (size_t)RQ_N * RQ_IN;

  // --- prep ---
  rq_zero_f32<<<4096, 256, 0, stream>>>(qf, (size_t)RQ_N * RQ_D);
  rq_zero_f32<<<1, 32, 0, stream>>>(acc, 2);

  rq_transpose_f32_to_f16<<<(RQ_IN*RQ_D + 255)/256, 256, 0, stream>>>(W1,  w1t,  RQ_IN, RQ_D);
  rq_transpose_f32_to_f16<<<(RQ_D*RQ_D  + 255)/256, 256, 0, stream>>>(W2,  w2t,  RQ_D,  RQ_D);
  rq_transpose_f32_to_f16<<<(RQ_D*RQ_IN + 255)/256, 256, 0, stream>>>(Wd1, wd1t, RQ_D,  RQ_IN);
  rq_transpose_f32_to_f16<<<(RQ_IN*RQ_IN+ 255)/256, 256, 0, stream>>>(Wd2, wd2t, RQ_IN, RQ_IN);
  rq_convert_f32_to_f16<<<4096, 256, 0, stream>>>(cb, cb16, (size_t)RQ_LEVELS*RQ_K*RQ_D);
  rq_cnorm<<<(RQ_LEVELS*RQ_K + 255)/256, 256, 0, stream>>>(cb, cn);

  // --- encoder ---
  rq_layernorm<<<RQ_N, 256, 0, stream>>>(x, ln_g, ln_b, xn);

  dim3 blk(32, 4);
  // enc GEMM1: [N,768] x [768,256], streaming A
  rq_wmma_gemm_stream<true ><<<dim3(RQ_N/16, RQ_D/128), blk, 0, stream>>>(
      xn, w1t, b1, nullptr, h1, RQ_IN, RQ_D);
  // enc GEMM2: [N,256] x [256,256], A-resident
  rq_wmma_gemm_ares<false><<<dim3(RQ_N/64), blk, 0, stream>>>(
      h1, w2t, b2, res, r16, RQ_D);

  // --- residual quantization, 4 levels ---
  for (int l = 0; l < RQ_LEVELS; ++l) {
    rq_quant_level<<<RQ_N/64, blk, 0, stream>>>(
        r16,
        cb16 + (size_t)l * RQ_K * RQ_D,
        cn   + (size_t)l * RQ_K,
        cb   + (size_t)l * RQ_K * RQ_D,
        res, qf, qf16, r16, out_idx, acc /*[0]=commit*/, l);
  }

  // --- decoder (decoder hidden reuses xn region) ---
  // dec GEMM1: [N,256] x [256,768], A-resident
  rq_wmma_gemm_ares<true ><<<dim3(RQ_N/64), blk, 0, stream>>>(
      qf16, wd1t, bd1, nullptr, xn, RQ_IN);
  // dec GEMM2: [N,768] x [768,768], streaming A
  rq_wmma_gemm_stream<false><<<dim3(RQ_N/16, RQ_IN/128), blk, 0, stream>>>(
      xn, wd2t, bd2, recon, nullptr, RQ_IN, RQ_IN);

  // --- losses ---
  rq_recloss<<<2048, 256, 0, stream>>>(recon, x, acc + 1, (size_t)RQ_N * RQ_IN);
  rq_finalize<<<1, 32, 0, stream>>>(acc, out);
}
